// InfiniAttention_21019569946670
// MI455X (gfx1250) — matrix operand order repaired and checked by reading, not compiled, not run on hardware
//
#include <hip/hip_runtime.h>
#include <math.h>

// Problem constants (from reference)
static constexpr int BB = 2, HH = 32, SS = 16384, DD = 64;
static constexpr int BH = BB * HH;         // 64 heads
static constexpr int CHUNK = 1024;         // seq rows per phase-1 block
static constexpr int ROWS  = 128;          // seq rows per phase-2 block

typedef _Float16 v2h  __attribute__((ext_vector_type(2)));
typedef _Float16 v8h  __attribute__((ext_vector_type(8)));
typedef _Float16 v16h __attribute__((ext_vector_type(16)));
typedef float    v8f  __attribute__((ext_vector_type(8)));

// Build a 16-half WMMA operand from two aligned 16-byte LDS reads (ds_load_b128).
__device__ __forceinline__ v16h ld_pair(const _Float16* p0, const _Float16* p1) {
  v8h a = *(const v8h*)p0;
  v8h b = *(const v8h*)p1;
  return __builtin_shufflevector(a, b, 0,1,2,3,4,5,6,7,8,9,10,11,12,13,14,15);
}

__device__ __forceinline__ float elu1(float x) {
  return x > 0.0f ? x + 1.0f : __expf(x);   // elu(x)+1 == exp(x) for x<=0
}

// ---------------------------------------------------------------------------
// Phase 1: per (bh, chunk) block accumulate  memory += k_feat^T @ V  (64x64)
//          and norm += sum_s k_feat  via f32 global atomics into workspace.
// ---------------------------------------------------------------------------
__global__ __launch_bounds__(256) void infini_phase1(
    const float* __restrict__ K, const float* __restrict__ V,
    float* __restrict__ wsM, float* __restrict__ wsN)
{
  const int bh    = blockIdx.x;
  const int chunk = blockIdx.y;
  const int t     = threadIdx.x;
  const int lane  = t & 31;
  const int wave  = t >> 5;          // 0..7
  const int half  = lane >> 4;       // hi/lo 16-lane group
  const int l15   = lane & 15;

  // Transposed f16 tiles: [feature 0..63][s-local 0..31], stride 40 halves (80B, 16B-aligned)
  __shared__ alignas(16) _Float16 ka[64 * 40];
  __shared__ alignas(16) _Float16 vb[64 * 40];
  __shared__ float nshare[64];

  if (t < 64) nshare[t] = 0.0f;

  const int d = t & 63;              // this thread's feature column
  const int g = t >> 6;              // 0..3 staging group
  const size_t base = ((size_t)bh * SS + (size_t)chunk * CHUNK) * DD;

  // wave -> two 16x16 output tiles: row tm, cols tn0, tn0+1 of the 4x4 grid
  const int tm  = wave >> 1;
  const int tn0 = (wave & 1) * 2;

  v8f acc0 = {}; v8f acc1 = {};
  float nsum = 0.0f;

  for (int step = 0; step < CHUNK / 32; ++step) {
    __syncthreads();
    const size_t sbase = base + (size_t)step * 32 * DD;

    // Stage 32x64 K and V tiles: convert + transpose into LDS
#pragma unroll
    for (int i = 0; i < 4; ++i) {
      const int p = g + 4 * i;       // s-pair index 0..15
      const int s = 2 * p;
      const float k0 = K[sbase + (size_t)s * DD + d];
      const float k1 = K[sbase + (size_t)(s + 1) * DD + d];
      const float w0 = V[sbase + (size_t)s * DD + d];
      const float w1 = V[sbase + (size_t)(s + 1) * DD + d];
      const float kf0 = elu1(k0), kf1 = elu1(k1);
      nsum += kf0 + kf1;
      v2h kp = { (_Float16)kf0, (_Float16)kf1 };
      v2h vp = { (_Float16)w0,  (_Float16)w1  };
      *(v2h*)&ka[d * 40 + s] = kp;
      *(v2h*)&vb[d * 40 + s] = vp;
    }
    // Hint the next 32x64 tiles into cache (global_prefetch_b8)
    if (step + 1 < CHUNK / 32) {
      __builtin_prefetch(&K[sbase + 32 * DD + (size_t)t * 32], 0, 1);
      __builtin_prefetch(&V[sbase + 32 * DD + (size_t)t * 32], 0, 1);
    }
    __syncthreads();

    // A operand: k_feat^T 16x32 tile, lane = M (feature), lo lanes K0..7|16..23, hi 8..15|24..31
    const int am = tm * 16 + l15;
    v16h A = ld_pair(&ka[am * 40 + half * 8], &ka[am * 40 + half * 8 + 16]);
    // B operands: V 32x16 tiles, lane = N (value dim), lo lanes K0..7|16..23, hi 8..15|24..31
    const int bn0 = tn0 * 16 + l15;
    v16h B0 = ld_pair(&vb[bn0 * 40 + half * 8], &vb[bn0 * 40 + half * 8 + 16]);
    v16h B1 = ld_pair(&vb[(bn0 + 16) * 40 + half * 8], &vb[(bn0 + 16) * 40 + half * 8 + 16]);

    acc0 = __builtin_amdgcn_wmma_f32_16x16x32_f16(false, A, false, B0, (short)0, acc0, false, false);
    acc1 = __builtin_amdgcn_wmma_f32_16x16x32_f16(false, A, false, B1, (short)0, acc1, false, false);
  }

  // Reduce norm within block, then atomically into workspace
  atomicAdd(&nshare[d], nsum);
  __syncthreads();
  if (t < 64) atomicAdd(&wsN[bh * 64 + t], nshare[t]);

  // Accumulate memory-matrix partials (C layout: VGPR r -> rows r / r+8)
  float* Mb = wsM + (size_t)bh * 4096;
#pragma unroll
  for (int r = 0; r < 8; ++r) {
    const int row = tm * 16 + r + half * 8;
    atomicAdd(&Mb[row * 64 + tn0 * 16 + l15],        acc0[r]);
    atomicAdd(&Mb[row * 64 + (tn0 + 1) * 16 + l15],  acc1[r]);
  }
}

// ---------------------------------------------------------------------------
// Phase 2: out = (q_feat @ memory) * (1 / (q_feat . norm)) per 128-row block.
// ---------------------------------------------------------------------------
__global__ __launch_bounds__(256) void infini_phase2(
    const float* __restrict__ Q, const float* __restrict__ wsM,
    const float* __restrict__ wsN, float* __restrict__ Out)
{
  const int bh   = blockIdx.x;
  const int rb   = blockIdx.y;
  const int t    = threadIdx.x;
  const int lane = t & 31;
  const int wave = t >> 5;
  const int half = lane >> 4;
  const int l15  = lane & 15;

  __shared__ alignas(16) _Float16 qf[ROWS * 72];  // [s-local][k], stride 72 halves (144B)
  __shared__ alignas(16) _Float16 mt[64 * 72];    // [v][k]  (B-operand layout for memory)
  __shared__ float nl[64];
  __shared__ float rden[ROWS];

  const float* Mb = wsM + (size_t)bh * 4096;
  // Stage memory matrix transposed into LDS as f16
#pragma unroll
  for (int i = 0; i < 16; ++i) {
    const int idx = t + 256 * i;     // idx = k*64 + v
    const int k = idx >> 6, v = idx & 63;
    mt[v * 72 + k] = (_Float16)Mb[idx];
  }
  if (t < 64) nl[t] = wsN[bh * 64 + t];

  // Stage 128x64 q_feat tile (row-major already matches A layout)
  const size_t qbase = ((size_t)bh * SS + (size_t)rb * ROWS) * DD;
#pragma unroll 8
  for (int i = 0; i < 32; ++i) {
    const int idx = t + 256 * i;     // idx = r*64 + k
    const int r = idx >> 6, k = idx & 63;
    qf[r * 72 + k] = (_Float16)elu1(Q[qbase + idx]);
  }
  __syncthreads();

  // Reciprocal denominators (f32 VALU dot, 64 terms)
  if (t < ROWS) {
    float den = 0.0f;
#pragma unroll
    for (int k = 0; k < 64; ++k) den += (float)qf[t * 72 + k] * nl[k];
    rden[t] = 1.0f / den;
  }

  // Each wave: rows [wave*16, wave*16+16), 4 output tiles across v, K = 64 (2 steps)
  const int row = wave * 16 + l15;
  v16h A0 = ld_pair(&qf[row * 72 + half * 8],      &qf[row * 72 + half * 8 + 16]);
  v16h A1 = ld_pair(&qf[row * 72 + 32 + half * 8], &qf[row * 72 + 32 + half * 8 + 16]);

  v8f acc[4];
#pragma unroll
  for (int tn = 0; tn < 4; ++tn) {
    const int v = tn * 16 + l15;
    v16h B0 = ld_pair(&mt[v * 72 + half * 8],      &mt[v * 72 + half * 8 + 16]);
    v16h B1 = ld_pair(&mt[v * 72 + 32 + half * 8], &mt[v * 72 + 32 + half * 8 + 16]);
    v8f c = {};
    c = __builtin_amdgcn_wmma_f32_16x16x32_f16(false, A0, false, B0, (short)0, c, false, false);
    c = __builtin_amdgcn_wmma_f32_16x16x32_f16(false, A1, false, B1, (short)0, c, false, false);
    acc[tn] = c;
  }
  __syncthreads();  // rden ready

  float* Ob = Out + qbase;
#pragma unroll
  for (int tn = 0; tn < 4; ++tn) {
#pragma unroll
    for (int r = 0; r < 8; ++r) {
      const int rr = wave * 16 + r + half * 8;
      Ob[(size_t)rr * DD + tn * 16 + l15] = acc[tn][r] * rden[rr];
    }
  }
}

extern "C" void kernel_launch(void* const* d_in, const int* in_sizes, int n_in,
                              void* d_out, int out_size, void* d_ws, size_t ws_size,
                              hipStream_t stream) {
  (void)in_sizes; (void)n_in; (void)out_size; (void)ws_size;
  const float* Q = (const float*)d_in[0];
  const float* K = (const float*)d_in[1];
  const float* V = (const float*)d_in[2];
  float* Out = (float*)d_out;

  float* wsM = (float*)d_ws;                       // [BH][64][64]
  float* wsN = wsM + (size_t)BH * 4096;            // [BH][64]
  hipMemsetAsync(d_ws, 0, (size_t)BH * (4096 + 64) * sizeof(float), stream);

  dim3 g1(BH, SS / CHUNK);   // 64 x 16
  infini_phase1<<<g1, 256, 0, stream>>>(K, V, wsM, wsN);

  dim3 g2(BH, SS / ROWS);    // 64 x 128
  infini_phase2<<<g2, 256, 0, stream>>>(Q, wsM, wsN, Out);
}
